// MultiSSA_73985106641104
// MI455X (gfx1250) — matrix operand order repaired and checked by reading, not compiled, not run on hardware
//
#include <hip/hip_runtime.h>

// ---------------- types ----------------
typedef _Float16 v16h __attribute__((ext_vector_type(16)));
typedef _Float16 v8h  __attribute__((ext_vector_type(8)));
typedef float    v8f  __attribute__((ext_vector_type(8)));

#define NB   8      // batch
#define NP   2048   // points
#define NC   128    // channels
#define NCO  256    // out channels
#define NS   512    // centers
#define NNEI 32     // n_near
#define NF   7      // branches
#define M4   (NB*NS)   // 4096 rows per branch

// ---------------- WMMA fragment loaders (ISA 7.12.2 layouts) ----------------
__device__ __forceinline__ v16h load_a_frag(const _Float16* row, int kk, int hi) {
    // lane(r,hi): K = {kk+hi*8 .. +7} U {kk+16+hi*8 .. +7}
    v8h a0 = *(const v8h*)(row + kk + hi * 8);
    v8h a1 = *(const v8h*)(row + kk + 16 + hi * 8);
    v16h a;
#pragma unroll
    for (int i = 0; i < 8; ++i) { a[i] = a0[i]; a[8 + i] = a1[i]; }
    return a;
}
__device__ __forceinline__ v16h load_b_frag(const _Float16* row, int kk, int hi) {
    // lane(r,hi) holds column n=r, K = {kk+hi*8 .. +7} U {kk+16+hi*8 .. +7}
    v8h b0 = *(const v8h*)(row + kk + hi * 8);
    v8h b1 = *(const v8h*)(row + kk + 16 + hi * 8);
    v16h b;
#pragma unroll
    for (int i = 0; i < 8; ++i) { b[i] = b0[i]; b[8 + i] = b1[i]; }
    return b;
}

// ---------------- elementwise converts ----------------
__global__ void f32_to_f16_k(const float* __restrict__ in, _Float16* __restrict__ out, size_t n) {
    size_t i = (size_t)blockIdx.x * blockDim.x + threadIdx.x;
    if (i < n) out[i] = (_Float16)in[i];
}

// out[f][n*K+k] = in[f][k*N+n]   (weight transpose + f32->f16)
__global__ void transpose_f16_k(const float* __restrict__ in, _Float16* __restrict__ out, int K, int N) {
    int f = blockIdx.z;
    size_t idx = (size_t)blockIdx.x * blockDim.x + threadIdx.x;
    size_t tot = (size_t)K * N;
    if (idx >= tot) return;
    int k = (int)(idx % K);
    int n = (int)(idx / K);
    out[(size_t)f * tot + (size_t)n * K + k] = (_Float16)in[(size_t)f * tot + (size_t)k * N + n];
}

// ---------------- row sum of squares ----------------
__global__ void rowsumsq_k(const float* __restrict__ x, float* __restrict__ xx) {
    int row = blockIdx.x;           // NB*NP rows
    int tid = threadIdx.x;          // 128
    __shared__ float s[128];
    float v = x[(size_t)row * NC + tid];
    s[tid] = v * v;
    __syncthreads();
    for (int st = 64; st > 0; st >>= 1) {
        if (tid < st) s[tid] += s[tid + st];
        __syncthreads();
    }
    if (tid == 0) xx[row] = s[0];
}

// ---------------- pairwise distance via WMMA gram ----------------
// d[b,i,j] = xx[i] + xx[j] - 2 * dot(x_i, x_j)
__global__ void gram_dist_k(const _Float16* __restrict__ xb, const float* __restrict__ xx,
                            float* __restrict__ d) {
    int b   = blockIdx.z;
    int tid = threadIdx.x;
    int lane = tid & 31, w = tid >> 5;
    int r = lane & 15, hi = lane >> 4;
    int i0 = (blockIdx.y * 8 + w) * 16;
    int j0 = blockIdx.x * 16;
    const _Float16* xB   = xb + (size_t)b * NP * NC;
    const _Float16* arow = xB + (size_t)(i0 + r) * NC;
    const _Float16* brow = xB + (size_t)(j0 + r) * NC;   // B col n = row j0+n
    v8f c = {};
#pragma unroll
    for (int kk = 0; kk < NC; kk += 32) {
        v16h a = load_a_frag(arow, kk, hi);
        v16h bb = load_b_frag(brow, kk, hi);
        c = __builtin_amdgcn_wmma_f32_16x16x32_f16(false, a, false, bb, (short)0, c, false, false);
    }
    int j = j0 + r;
    float xxj = xx[(size_t)b * NP + j];
#pragma unroll
    for (int v = 0; v < 8; ++v) {
        int i = i0 + v + 8 * hi;
        d[((size_t)b * NP + i) * NP + j] = xx[(size_t)b * NP + i] + xxj - 2.0f * c[v];
    }
}

// ---------------- top-11 nearest per row (adjacency seeds) ----------------
__global__ void topk11_k(const float* __restrict__ d, int* __restrict__ nbr) {
    int row = blockIdx.x;                 // NB*NP
    int tid = threadIdx.x;                // 256
    const float* drow = d + (size_t)row * NP;
    float vals[8];
#pragma unroll
    for (int e = 0; e < 8; ++e) vals[e] = drow[tid + e * 256];
    __shared__ float sv[256];
    __shared__ int   si[256];
    for (int sel = 0; sel < 11; ++sel) {
        float bv = INFINITY; int bi = 0x7fffffff;
#pragma unroll
        for (int e = 0; e < 8; ++e) {
            int j = tid + e * 256;
            if (vals[e] < bv) { bv = vals[e]; bi = j; }
        }
        sv[tid] = bv; si[tid] = bi;
        __syncthreads();
        for (int st = 128; st > 0; st >>= 1) {
            if (tid < st) {
                if (sv[tid + st] < sv[tid] ||
                    (sv[tid + st] == sv[tid] && si[tid + st] < si[tid])) {
                    sv[tid] = sv[tid + st]; si[tid] = si[tid + st];
                }
            }
            __syncthreads();
        }
        int win = si[0];
        if (tid == 0) nbr[(size_t)row * 11 + sel] = win;
        if ((win & 255) == tid) vals[win >> 8] = INFINITY;
        __syncthreads();
    }
}

// ---------------- adjacency bitmasks (64 u32 per row) ----------------
__global__ void build_adj_k(const int* __restrict__ nbr, unsigned* __restrict__ adj) {
    int row = blockIdx.x * blockDim.x + threadIdx.x;
    if (row >= NB * NP) return;
    unsigned* a = adj + (size_t)row * 64;
#pragma unroll
    for (int w = 0; w < 64; ++w) a[w] = 0u;
    for (int t = 0; t < 11; ++t) {
        int j = nbr[(size_t)row * 11 + t];
        a[j >> 5] |= 1u << (j & 31);
    }
}

// ---------------- farthest point sampling (sequential scan) ----------------
__global__ void fps_k(const float* __restrict__ d, int* __restrict__ fpsidx) {
    int b = blockIdx.x;                   // NB
    int tid = threadIdx.x;                // 256
    __shared__ float mind[NP];
    __shared__ float sv[256];
    __shared__ int   si[256];
    __shared__ int   sprev;
#pragma unroll
    for (int e = 0; e < 8; ++e) mind[tid + e * 256] = INFINITY;
    if (tid == 0) { fpsidx[(size_t)b * NS] = 0; sprev = 0; }
    __syncthreads();
    for (int it = 1; it < NS; ++it) {
        int prev = sprev;
        const float* drow = d + ((size_t)b * NP + prev) * NP;
#pragma unroll
        for (int e = 0; e < 8; ++e) {
            int j = tid + e * 256;
            mind[j] = fminf(mind[j], drow[j]);
        }
        __syncthreads();
        float bv = -INFINITY; int bi = 0;
#pragma unroll
        for (int e = 0; e < 8; ++e) {
            int j = tid + e * 256;
            float v = mind[j];
            if (v > bv) { bv = v; bi = j; }
        }
        sv[tid] = bv; si[tid] = bi;
        __syncthreads();
        for (int st = 128; st > 0; st >>= 1) {
            if (tid < st) {
                if (sv[tid + st] > sv[tid] ||
                    (sv[tid + st] == sv[tid] && si[tid + st] < si[tid])) {
                    sv[tid] = sv[tid + st]; si[tid] = si[tid + st];
                }
            }
            __syncthreads();
        }
        if (tid == 0) { fpsidx[(size_t)b * NS + it] = si[0]; sprev = si[0]; }
        __syncthreads();
    }
}

// ---------------- 2-hop reach (bitmask BFS) + top-32 on FPS rows ----------------
__global__ void reach_top32_k(const float* __restrict__ d, const unsigned* __restrict__ adj,
                              const int* __restrict__ fpsidx, int* __restrict__ kidx) {
    int bs = blockIdx.x;                  // NB*NS
    int b  = bs >> 9;
    int tid = threadIdx.x;                // 256
    __shared__ unsigned reach[64], nreach[64];
    __shared__ float sv[256];
    __shared__ int   si[256];
    __shared__ int   srow;
    if (tid == 0) srow = fpsidx[bs];
    __syncthreads();
    int i = srow;
    const unsigned* adjb = adj + (size_t)b * NP * 64;
    if (tid < 64) reach[tid] = adjb[(size_t)i * 64 + tid];
    __syncthreads();
    for (int hop = 0; hop < 2; ++hop) {
        if (tid < 64) nreach[tid] = reach[tid];
        __syncthreads();
        int w = tid & 63, part = tid >> 6;    // 4 partitions x 64 words
        unsigned acc = 0u;
        for (int j = part * 512; j < part * 512 + 512; ++j) {
            if ((reach[j >> 5] >> (j & 31)) & 1u)
                acc |= adjb[(size_t)j * 64 + w];
        }
        atomicOr(&nreach[w], acc);
        __syncthreads();
        if (tid < 64) reach[tid] = nreach[tid];
        __syncthreads();
    }
    const float* drow = d + ((size_t)b * NP + i) * NP;
    float vals[8];
#pragma unroll
    for (int e = 0; e < 8; ++e) {
        int j = tid + e * 256;
        bool rb = (reach[j >> 5] >> (j & 31)) & 1u;
        vals[e] = rb ? drow[j] : INFINITY;
    }
    __syncthreads();
    for (int sel = 0; sel < NNEI; ++sel) {
        float bv = INFINITY; int bi = 0x7fffffff;
#pragma unroll
        for (int e = 0; e < 8; ++e) {
            int j = tid + e * 256;
            if (vals[e] < bv) { bv = vals[e]; bi = j; }
        }
        sv[tid] = bv; si[tid] = bi;
        __syncthreads();
        for (int st = 128; st > 0; st >>= 1) {
            if (tid < st) {
                if (sv[tid + st] < sv[tid] ||
                    (sv[tid + st] == sv[tid] && si[tid + st] < si[tid])) {
                    sv[tid] = sv[tid + st]; si[tid] = si[tid + st];
                }
            }
            __syncthreads();
        }
        int win = si[0];
        if (tid == 0) kidx[(size_t)bs * NNEI + sel] = win;
        if ((win & 255) == tid) vals[win >> 8] = INFINITY;
        __syncthreads();
    }
}

// ---------------- gather centers (f16) ----------------
__global__ void gather_centers_k(const _Float16* __restrict__ featb, const int* __restrict__ fpsidx,
                                 _Float16* __restrict__ cent) {
    size_t idx = (size_t)blockIdx.x * blockDim.x + threadIdx.x;
    size_t total = (size_t)NF * M4 * NC;
    if (idx >= total) return;
    int c = (int)(idx & (NC - 1));
    size_t t = idx >> 7;
    int m = (int)(t % M4);
    int f = (int)(t / M4);
    int b = m >> 9;
    int src = fpsidx[m];
    cent[idx] = featb[(((size_t)f * NB + b) * NP + src) * NC + c];
}

// ---------------- generic batched f16 WMMA GEMM:  out = X @ Wb^T + bias ----------------
// X: (NF, M, K) f16 row-major.  Wb: (NF, N, K) f16 (i.e. math-B transposed; B(k,n)=Wb[n*K+k]).
// bias: (NF, N) f32 or null.  outF: (NF, M, N) f32 or null.  outH: same in f16 or null.
__global__ void gemm_f16_k(const _Float16* __restrict__ X, const _Float16* __restrict__ Wb,
                           const float* __restrict__ bias, float* __restrict__ outF,
                           _Float16* __restrict__ outH, int M, int K, int N, int relu) {
    int f   = blockIdx.z;
    int tid = threadIdx.x;
    int lane = tid & 31, w = tid >> 5;
    int r = lane & 15, hi = lane >> 4;
    int m0 = (blockIdx.y * 8 + w) * 16;
    int n0 = blockIdx.x * 16;
    if (m0 >= M) return;
    const _Float16* arow = X  + (size_t)f * M * K + (size_t)(m0 + r) * K;
    const _Float16* brow = Wb + (size_t)f * N * K + (size_t)(n0 + r) * K;
    v8f c = {};
    for (int kk = 0; kk < K; kk += 32) {
        v16h a = load_a_frag(arow, kk, hi);
        v16h b = load_b_frag(brow, kk, hi);
        c = __builtin_amdgcn_wmma_f32_16x16x32_f16(false, a, false, b, (short)0, c, false, false);
    }
    int col = n0 + r;
    float bv = bias ? bias[(size_t)f * N + col] : 0.0f;
#pragma unroll
    for (int v = 0; v < 8; ++v) {
        int mrow = m0 + v + 8 * hi;
        float val = c[v] + bv;
        if (relu) val = fmaxf(val, 0.0f);
        size_t o = ((size_t)f * M + mrow) * N + col;
        if (outF) outF[o] = val;
        if (outH) outH[o] = (_Float16)val;
    }
}

// ---------------- attention core: scores, softmax, attn-weighted feature mix ----------------
// score_t = (g_t . (wk@q) + q.bk) / sqrt(C);  Gt = softmax(score) @ G
__global__ void attn_core_k(const _Float16* __restrict__ featb, const float* __restrict__ Qf,
                            const float* __restrict__ QHf, const float* __restrict__ ab_k,
                            const int* __restrict__ kidx, _Float16* __restrict__ Gt) {
    int m = blockIdx.x;        // M4
    int f = blockIdx.y;        // NF
    int tid = threadIdx.x;     // 128
    int b = m >> 9;
    __shared__ float qh[NC];
    __shared__ float red[NC];
    __shared__ float sc[NNEI];
    size_t qoff = ((size_t)f * M4 + m) * NC;
    qh[tid]  = QHf[qoff + tid];
    red[tid] = Qf[qoff + tid] * ab_k[(size_t)f * NC + tid];
    __syncthreads();
    for (int st = 64; st > 0; st >>= 1) {
        if (tid < st) red[tid] += red[tid + st];
        __syncthreads();
    }
    float s0 = red[0];
    const _Float16* fb = featb + ((size_t)f * NB + b) * NP * NC;
    const int* ki = kidx + (size_t)m * NNEI;
    if (tid < NNEI) {
        int j = ki[tid];
        const _Float16* g = fb + (size_t)j * NC;
        float acc = 0.0f;
        for (int c2 = 0; c2 < NC; ++c2) acc += (float)g[c2] * qh[c2];
        sc[tid] = (acc + s0) * 0.088388347648318447f;   // 1/sqrt(128)
    }
    __syncthreads();
    if (tid == 0) {
        float mx = sc[0];
        for (int t = 1; t < NNEI; ++t) mx = fmaxf(mx, sc[t]);
        float sm = 0.0f;
        for (int t = 0; t < NNEI; ++t) { sc[t] = __expf(sc[t] - mx); sm += sc[t]; }
        float inv = 1.0f / sm;
        for (int t = 0; t < NNEI; ++t) sc[t] *= inv;
    }
    __syncthreads();
    float acc = 0.0f;
    for (int t = 0; t < NNEI; ++t) {
        int j = ki[t];
        acc += sc[t] * (float)fb[(size_t)j * NC + tid];
    }
    Gt[qoff + tid] = (_Float16)acc;
}

// ---------------- host launcher ----------------
extern "C" void kernel_launch(void* const* d_in, const int* in_sizes, int n_in,
                              void* d_out, int out_size, void* d_ws, size_t ws_size,
                              hipStream_t stream) {
    (void)in_sizes; (void)n_in; (void)out_size; (void)ws_size;
    const float* feats[NF];
    for (int f = 0; f < NF; ++f) feats[f] = (const float*)d_in[f];
    const float* aw_q = (const float*)d_in[7];
    const float* ab_q = (const float*)d_in[8];
    const float* aw_k = (const float*)d_in[9];
    const float* ab_k = (const float*)d_in[10];
    const float* aw_v = (const float*)d_in[11];
    const float* ab_v = (const float*)d_in[12];
    const float* mw1  = (const float*)d_in[13];
    const float* mb1  = (const float*)d_in[14];
    const float* mw2  = (const float*)d_in[15];
    const float* mb2  = (const float*)d_in[16];

    // workspace carve (256B aligned chunks)
    char* p = (char*)d_ws;
    auto carve = [&](size_t bytes) -> void* {
        void* r = (void*)p;
        p += (bytes + 255) & ~(size_t)255;
        return r;
    };
    const size_t FEAT_E = (size_t)NB * NP * NC;                     // per-feature elems
    _Float16* featb  = (_Float16*)carve(NF * FEAT_E * 2);
    _Float16* wqT    = (_Float16*)carve((size_t)NF * NC * NC * 2);
    _Float16* wk16   = (_Float16*)carve((size_t)NF * NC * NC * 2);
    _Float16* wvT    = (_Float16*)carve((size_t)NF * NC * NC * 2);
    _Float16* w1T    = (_Float16*)carve((size_t)NF * NC * NCO * 2);
    _Float16* w2T    = (_Float16*)carve((size_t)NF * NCO * NCO * 2);
    float*    xx     = (float*)carve((size_t)NB * NP * 4);
    float*    dmat   = (float*)carve((size_t)NB * NP * NP * 4);     // 134 MB
    int*      nbr    = (int*)carve((size_t)NB * NP * 11 * 4);
    unsigned* adj    = (unsigned*)carve((size_t)NB * NP * 64 * 4);
    int*      fpsidx = (int*)carve((size_t)NB * NS * 4);
    int*      kidx   = (int*)carve((size_t)NB * NS * NNEI * 4);
    _Float16* cent16 = (_Float16*)carve((size_t)NF * M4 * NC * 2);
    float*    Qf     = (float*)carve((size_t)NF * M4 * NC * 4);
    _Float16* Q16    = (_Float16*)carve((size_t)NF * M4 * NC * 2);
    float*    QHf    = (float*)carve((size_t)NF * M4 * NC * 4);
    _Float16* Gt16   = (_Float16*)carve((size_t)NF * M4 * NC * 2);
    _Float16* O16    = (_Float16*)carve((size_t)NF * M4 * NC * 2);
    _Float16* h16    = (_Float16*)carve((size_t)NF * M4 * NCO * 2);

    // 1) feature + weight converts
    for (int f = 0; f < NF; ++f)
        f32_to_f16_k<<<(unsigned)((FEAT_E + 255) / 256), 256, 0, stream>>>(feats[f], featb + f * FEAT_E, FEAT_E);
    {
        size_t n = (size_t)NF * NC * NC;
        f32_to_f16_k<<<(unsigned)((n + 255) / 256), 256, 0, stream>>>(aw_k, wk16, n);  // plain: B(k,n)=wk[n,k]
    }
    transpose_f16_k<<<dim3((NC * NC + 255) / 256, 1, NF), 256, 0, stream>>>(aw_q, wqT, NC, NC);
    transpose_f16_k<<<dim3((NC * NC + 255) / 256, 1, NF), 256, 0, stream>>>(aw_v, wvT, NC, NC);
    transpose_f16_k<<<dim3((NC * NCO + 255) / 256, 1, NF), 256, 0, stream>>>(mw1, w1T, NC, NCO);
    transpose_f16_k<<<dim3((NCO * NCO + 255) / 256, 1, NF), 256, 0, stream>>>(mw2, w2T, NCO, NCO);

    // 2) pairwise distances (WMMA gram)
    rowsumsq_k<<<NB * NP, 128, 0, stream>>>(feats[0], xx);
    gram_dist_k<<<dim3(NP / 16, NP / 128, NB), 256, 0, stream>>>(featb /* f=0 = xyz */, xx, dmat);

    // 3) kNN graph + FPS + 2-hop surface kNN at FPS rows
    topk11_k<<<NB * NP, 256, 0, stream>>>(dmat, nbr);
    build_adj_k<<<(NB * NP + 255) / 256, 256, 0, stream>>>(nbr, adj);
    fps_k<<<NB, 256, 0, stream>>>(dmat, fpsidx);
    reach_top32_k<<<NB * NS, 256, 0, stream>>>(dmat, adj, fpsidx, kidx);

    // 4) gather centers
    {
        size_t tot = (size_t)NF * M4 * NC;
        gather_centers_k<<<(unsigned)((tot + 255) / 256), 256, 0, stream>>>(featb, fpsidx, cent16);
    }

    // 5) attention + MLP as batched WMMA GEMMs
    // Q = centers @ wq + bq                  (f32 + f16 outputs)
    gemm_f16_k<<<dim3(NC / 16, M4 / 128, NF), 256, 0, stream>>>(cent16, wqT, ab_q, Qf, Q16, M4, NC, NC, 0);
    // QH = Q @ wk^T  (qh = wk @ q), no bias  (f32 output)
    gemm_f16_k<<<dim3(NC / 16, M4 / 128, NF), 256, 0, stream>>>(Q16, wk16, nullptr, QHf, nullptr, M4, NC, NC, 0);
    // scores/softmax/attn-mix -> Gt = attn @ G  (f16)
    attn_core_k<<<dim3(M4, NF), 128, 0, stream>>>(featb, Qf, QHf, ab_k, kidx, Gt16);
    // O = Gt @ wv + bv                       (f16 output)
    gemm_f16_k<<<dim3(NC / 16, M4 / 128, NF), 256, 0, stream>>>(Gt16, wvT, ab_v, nullptr, O16, M4, NC, NC, 0);
    // h = relu(O @ w1 + b1)                  (f16 output)
    gemm_f16_k<<<dim3(NCO / 16, M4 / 128, NF), 256, 0, stream>>>(O16, w1T, mb1, nullptr, h16, M4, NC, NCO, 1);
    // out = relu(h @ w2 + b2)                (f32 -> d_out)
    gemm_f16_k<<<dim3(NCO / 16, M4 / 128, NF), 256, 0, stream>>>(h16, w2T, mb2, (float*)d_out, nullptr, M4, NCO, NCO, 1);
}
